// AttenderAggregator_90331752170181
// MI455X (gfx1250) — compile-verified, matrix operand order checked by reading, not run on hardware
//
#include <hip/hip_runtime.h>
#include <hip/hip_bf16.h>
#include <stdint.h>

typedef _Float16 f16;
typedef __attribute__((ext_vector_type(16))) _Float16 v16h;
typedef __attribute__((ext_vector_type(8)))  float    v8f;

union Frag16 { v16h v; uint4 q[2]; };

// ---- problem dims ----
#define NB      16
#define RB      64
#define HTB     24
#define NRB     (NB*RB)          // 1024
#define MROWS   (NRB*HTB)        // 24576
#define DM      256
#define RELD    808
#define RELP    832              // pad to mult of 32
#define QKVN    2424
#define QKVP    2432
#define DFFN    3232
#define DFFP    3264
#define HIDN    512
#define NRELN   97
#define NRELP   128
#define NHEADS  8
#define HD      101

// LDS byte-offset of a __shared__ pointer (addrspace(3) cast strips the aperture)
typedef __attribute__((address_space(3))) const f16 lds_cf16;
__device__ __forceinline__ uint32_t lds_off_u32(const f16* p) {
  return (uint32_t)(uintptr_t)(lds_cf16*)p;
}

// CDNA5 async DMA: memory -> LDS, 16B per lane, tracked by ASYNCcnt
#define ASYNC_CP_B128(ldsoff, gaddr)                                           \
  asm volatile("global_load_async_to_lds_b128 %0, %1, off"                     \
               :: "v"(ldsoff), "v"(gaddr) : "memory")
#define WAIT_ASYNC() asm volatile("s_wait_asynccnt 0x0" ::: "memory")

// =====================================================================
// WMMA GEMM: out = act(A[M x K] * W[N x K]^T + bias)
// A f16 row-major (lda), W f16 row-major [Npad x Kpad] (ldb), K mult of 32.
// Block = 256 thr (8 waves). Tile 128(M) x 64(N), BK=32, double-buffered
// LDS filled by global_load_async_to_lds_b128. Each wave: 32x32 = 4 WMMAs.
// =====================================================================
__global__ __launch_bounds__(256) void gemm_wmma(
    const f16* __restrict__ A, int lda,
    const f16* __restrict__ B, int ldb,
    const float* __restrict__ bias,
    f16* __restrict__ outh, int ldh,
    float* __restrict__ outf, int ldf,
    int Nreal, int K, int relu)
{
  __shared__ f16 As[2][128 * 32];   // 2 x 8KB
  __shared__ f16 Bs[2][64 * 32];    // 2 x 4KB
  const int tid   = threadIdx.x;
  const int lane  = tid & 31;
  const int wave  = tid >> 5;
  const int waveM = wave >> 1;      // 0..3 -> 32 rows each
  const int waveN = wave & 1;       // 0..1 -> 32 cols each
  const int bM = blockIdx.y * 128;
  const int bN = blockIdx.x * 64;

  // staging assignment: thread covers A rows r0 and r0+64, B row r0 (8 halves each)
  const int r0 = tid >> 2;           // 0..63
  const int ch = (tid & 3) * 8;      // half offset within BK
  const unsigned long long pA0 =
      (unsigned long long)(uintptr_t)(A + (size_t)(bM + r0) * lda + ch);
  const unsigned long long pA1 =
      (unsigned long long)(uintptr_t)(A + (size_t)(bM + r0 + 64) * lda + ch);
  const unsigned long long pB =
      (unsigned long long)(uintptr_t)(B + (size_t)(bN + r0) * ldb + ch);
  const uint32_t lA0[2] = { lds_off_u32(&As[0][r0 * 32 + ch]),
                            lds_off_u32(&As[1][r0 * 32 + ch]) };
  const uint32_t lA1[2] = { lds_off_u32(&As[0][(r0 + 64) * 32 + ch]),
                            lds_off_u32(&As[1][(r0 + 64) * 32 + ch]) };
  const uint32_t lB[2]  = { lds_off_u32(&Bs[0][r0 * 32 + ch]),
                            lds_off_u32(&Bs[1][r0 * 32 + ch]) };

  // fragment read offsets
  const int kb  = ((lane >> 4) & 1) * 8;                 // K sub-offset per half-wave
  const int arA = (waveM * 32 + (lane & 15)) * 32;       // a0 rows
  const int arB = arA + 16 * 32;                         // a1 rows
  const int brA = (waveN * 32 + (lane & 15)) * 32;       // b0 cols
  const int brB = brA + 16 * 32;                         // b1 cols

  v8f acc00 = {}, acc01 = {}, acc10 = {}, acc11 = {};

  const int nk = K >> 5;
  // prologue: stage tile 0 into buffer 0
  {
    ASYNC_CP_B128(lA0[0], pA0);
    ASYNC_CP_B128(lA1[0], pA1);
    ASYNC_CP_B128(lB[0],  pB);
    WAIT_ASYNC();
    __syncthreads();
  }
  for (int ki = 0; ki < nk; ++ki) {
    const int buf = ki & 1;
    if (ki + 1 < nk) {                       // prefetch next tile (other buffer)
      const unsigned long long o = (unsigned long long)((ki + 1) << 5) * 2;
      ASYNC_CP_B128(lA0[buf ^ 1], pA0 + o);
      ASYNC_CP_B128(lA1[buf ^ 1], pA1 + o);
      ASYNC_CP_B128(lB[buf ^ 1],  pB  + o);
    }
    const f16* Ab = &As[buf][0];
    const f16* Bb = &Bs[buf][0];
    Frag16 a0, a1, b0, b1;
    a0.q[0] = *(const uint4*)(&Ab[arA + kb]);
    a0.q[1] = *(const uint4*)(&Ab[arA + kb + 16]);
    a1.q[0] = *(const uint4*)(&Ab[arB + kb]);
    a1.q[1] = *(const uint4*)(&Ab[arB + kb + 16]);
    b0.q[0] = *(const uint4*)(&Bb[brA + kb]);
    b0.q[1] = *(const uint4*)(&Bb[brA + kb + 16]);
    b1.q[0] = *(const uint4*)(&Bb[brB + kb]);
    b1.q[1] = *(const uint4*)(&Bb[brB + kb + 16]);
    acc00 = __builtin_amdgcn_wmma_f32_16x16x32_f16(false, a0.v, false, b0.v,
                                                   (short)0, acc00, false, false);
    acc01 = __builtin_amdgcn_wmma_f32_16x16x32_f16(false, a0.v, false, b1.v,
                                                   (short)0, acc01, false, false);
    acc10 = __builtin_amdgcn_wmma_f32_16x16x32_f16(false, a1.v, false, b0.v,
                                                   (short)0, acc10, false, false);
    acc11 = __builtin_amdgcn_wmma_f32_16x16x32_f16(false, a1.v, false, b1.v,
                                                   (short)0, acc11, false, false);
    WAIT_ASYNC();            // next tile landed in LDS
    __syncthreads();         // all waves done reading buf & landing buf^1
  }

  // C/D layout: VGPR v -> M = v (+8 for lanes 16..31), N = lane&15 (+16 for *1)
  const int hi8 = ((lane >> 4) & 1) * 8;
  const int mb = bM + waveM * 32 + hi8;
  const int nb = bN + waveN * 32 + (lane & 15);
  v8f accs[2][2] = { { acc00, acc01 }, { acc10, acc11 } };
#pragma unroll
  for (int aT = 0; aT < 2; ++aT) {
#pragma unroll
    for (int bT = 0; bT < 2; ++bT) {
      const int col = nb + bT * 16;
      if (col >= Nreal) continue;
      const float bv = bias[col];
#pragma unroll
      for (int v = 0; v < 8; ++v) {
        const int row = mb + aT * 16 + v;
        float y = accs[aT][bT][v] + bv;
        if (relu) y = fmaxf(y, 0.f);
        if (outh) outh[(size_t)row * ldh + col] = (f16)y;
        if (outf) outf[(size_t)row * ldf + col] = y;
      }
    }
  }
}

// =====================================================================
// Build rel features: X[M x 808] = concat(ah, at, ah*at, h2t, t2h) * mask
// =====================================================================
__global__ __launch_bounds__(256) void build_rel(
    const float* __restrict__ mse,
    const int* __restrict__ hidx, const int* __restrict__ hmask,
    const int* __restrict__ tidx, const int* __restrict__ tmask,
    const int* __restrict__ htidx, const int* __restrict__ htmask,
    const int* __restrict__ d1, const int* __restrict__ d2,
    const float* __restrict__ dis_embed,
    float* __restrict__ X)
{
  const int m  = blockIdx.x;           // row in [0, MROWS)
  const int nr = m / HTB;
  const int n  = nr / RB;
  const int r  = nr % RB;
  const int hi = htidx[m * 2 + 0];
  const int ti = htidx[m * 2 + 1];
  const float fm  = (htmask[m] != 0) ? 1.f : 0.f;
  const int hb = (n * RB + r) * 8;
  const int hs = hidx[hb + hi];
  const int ts = tidx[hb + ti];
  const float fhm = (hmask[hb + hi] != 0) ? 1.f : 0.f;
  const float ftm = (tmask[hb + ti] != 0) ? 1.f : 0.f;
  float* xr = X + (size_t)m * RELD;
  const int e = threadIdx.x;           // 0..255 == DM
  const float a = mse[(size_t)(n * 32 + hs) * DM + e] * fhm;
  const float t = mse[(size_t)(n * 32 + ts) * DM + e] * ftm;
  xr[e]           = a * fm;
  xr[DM + e]      = t * fm;
  xr[2 * DM + e]  = a * t * fm;
  if (e < 20) {
    xr[3 * DM + e]      = dis_embed[d1[m] * 20 + e] * fm;
    xr[3 * DM + 20 + e] = dis_embed[d2[m] * 20 + e] * fm;
  }
}

// X_f32[M x 808] -> X_f16[M x 832] with zero K-pad
__global__ __launch_bounds__(256) void conv_x(const float* __restrict__ X,
                                              f16* __restrict__ Xh)
{
  const size_t idx = (size_t)blockIdx.x * 256 + threadIdx.x;  // < MROWS*RELP
  const int m = (int)(idx / RELP), k = (int)(idx % RELP);
  Xh[idx] = (k < RELD) ? (f16)X[(size_t)m * RELD + k] : (f16)0.f;
}

// W f32 [N x K] -> f16 [Npad x Kpad] zero-padded
__global__ __launch_bounds__(256) void conv_w(const float* __restrict__ W,
                                              f16* __restrict__ O,
                                              int N, int K, int Kp, long total)
{
  const long idx = (long)blockIdx.x * 256 + threadIdx.x;
  if (idx >= total) return;
  const int n = (int)(idx / Kp), k = (int)(idx % Kp);
  O[idx] = (n < N && k < K) ? (f16)W[(size_t)n * K + k] : (f16)0.f;
}

__global__ __launch_bounds__(256) void conv_b(const float* __restrict__ B,
                                              float* __restrict__ O, int N, int Np)
{
  const int idx = blockIdx.x * 256 + threadIdx.x;
  if (idx < Np) O[idx] = (idx < N) ? B[idx] : 0.f;
}

// =====================================================================
// MHA for one (nr, head): S=24, d=101, f32 math in LDS.
// =====================================================================
__global__ __launch_bounds__(256) void attn_kernel(
    const f16* __restrict__ QKV,          // [M x QKVP]
    const int* __restrict__ htmask,
    f16* __restrict__ O)                  // [M x RELP]
{
  __shared__ float q[HTB][HD + 1], kk[HTB][HD + 1], vv[HTB][HD + 1];
  __shared__ float sc[HTB][HTB];
  const int nr = blockIdx.x >> 3;
  const int h  = blockIdx.x & 7;
  const int tid = threadIdx.x;
  for (int idx = tid; idx < HTB * HD; idx += 256) {
    const int i = idx / HD, c = idx % HD;
    const size_t base = (size_t)(nr * HTB + i) * QKVP + h * HD + c;
    q[i][c]  = (float)QKV[base];
    kk[i][c] = (float)QKV[base + RELD];
    vv[i][c] = (float)QKV[base + 2 * RELD];
  }
  __syncthreads();
  const float scale = rsqrtf((float)HD);
  for (int p = tid; p < HTB * HTB; p += 256) {
    const int i = p / HTB, j = p % HTB;
    float s = 0.f;
    for (int c = 0; c < HD; ++c) s += q[i][c] * kk[j][c];
    s *= scale;
    if (htmask[nr * HTB + j] == 0) s = -1e9f;
    sc[i][j] = s;
  }
  __syncthreads();
  if (tid < HTB) {
    float mx = -1e30f;
    for (int j = 0; j < HTB; ++j) mx = fmaxf(mx, sc[tid][j]);
    float sum = 0.f;
    for (int j = 0; j < HTB; ++j) { float e = expf(sc[tid][j] - mx); sc[tid][j] = e; sum += e; }
    const float inv = 1.f / sum;
    for (int j = 0; j < HTB; ++j) sc[tid][j] *= inv;
  }
  __syncthreads();
  for (int idx = tid; idx < HTB * HD; idx += 256) {
    const int i = idx / HD, c = idx % HD;
    float o = 0.f;
    for (int j = 0; j < HTB; ++j) o += sc[i][j] * vv[j][c];
    O[(size_t)(nr * HTB + i) * RELP + h * HD + c] = (f16)o;
  }
}

// zero columns [808, 832) of O
__global__ __launch_bounds__(256) void zero_pad_o(f16* __restrict__ O)
{
  const int idx = blockIdx.x * 256 + threadIdx.x;   // < MROWS*24
  const int row = idx / 24, c = RELD + idx % 24;
  O[(size_t)row * RELP + c] = (f16)0.f;
}

// X = LN(X + Z) over 808 features (Z may be null)
__global__ __launch_bounds__(256) void ln_kernel(
    float* __restrict__ X, const float* __restrict__ Z,
    const float* __restrict__ g, const float* __restrict__ b)
{
  __shared__ float s1[256], s2[256];
  const int m = blockIdx.x;
  float* xr = X + (size_t)m * RELD;
  const float* zr = Z ? Z + (size_t)m * RELD : nullptr;
  float xs[4];
  float sum = 0.f, sq = 0.f;
  int j = 0;
  for (int i = threadIdx.x; i < RELD; i += 256, ++j) {
    float x = xr[i] + (zr ? zr[i] : 0.f);
    xs[j] = x; sum += x; sq += x * x;
  }
  s1[threadIdx.x] = sum; s2[threadIdx.x] = sq;
  __syncthreads();
  for (int off = 128; off > 0; off >>= 1) {
    if (threadIdx.x < off) {
      s1[threadIdx.x] += s1[threadIdx.x + off];
      s2[threadIdx.x] += s2[threadIdx.x + off];
    }
    __syncthreads();
  }
  const float mean = s1[0] * (1.f / RELD);
  const float var  = s2[0] * (1.f / RELD) - mean * mean;
  const float inv  = rsqrtf(var + 1e-5f);
  j = 0;
  for (int i = threadIdx.x; i < RELD; i += 256, ++j)
    xr[i] = (xs[j] - mean) * inv * g[i] + b[i];
}

// masked logsumexp over HT: out[nr*97+rel]
__global__ __launch_bounds__(128) void lse_kernel(
    const float* __restrict__ logits,     // [M x NRELP]
    const int* __restrict__ htmask,
    float* __restrict__ out)
{
  const int nr = blockIdx.x;
  const int rel = threadIdx.x;
  if (rel >= NRELN) return;
  float mx = -1e30f;
  for (int t = 0; t < HTB; ++t) {
    float x = logits[(size_t)(nr * HTB + t) * NRELP + rel];
    if (htmask[nr * HTB + t] == 0) x -= 1e10f;
    mx = fmaxf(mx, x);
  }
  float sum = 0.f;
  for (int t = 0; t < HTB; ++t) {
    float x = logits[(size_t)(nr * HTB + t) * NRELP + rel];
    if (htmask[nr * HTB + t] == 0) x -= 1e10f;
    sum += expf(x - mx);
  }
  out[(size_t)nr * NRELN + rel] = mx + logf(sum);
}

// =====================================================================
// Host orchestration
// =====================================================================
extern "C" void kernel_launch(void* const* d_in, const int* in_sizes, int n_in,
                              void* d_out, int out_size, void* d_ws, size_t ws_size,
                              hipStream_t stream) {
  (void)in_sizes; (void)n_in; (void)out_size; (void)ws_size;
  // ---- inputs (setup_inputs dict order; params flattened in insertion order) ----
  const float* mse    = (const float*)d_in[0];
  const int*   hidx   = (const int*)d_in[1];
  const int*   hmask  = (const int*)d_in[2];
  const int*   tidx   = (const int*)d_in[3];
  const int*   tmask  = (const int*)d_in[4];
  const int*   htidx  = (const int*)d_in[5];
  const int*   htmask = (const int*)d_in[6];
  const int*   d1     = (const int*)d_in[7];
  const int*   d2     = (const int*)d_in[8];
  const float* disemb = (const float*)d_in[9];
  const float* norm_g = (const float*)d_in[34];
  const float* norm_b = (const float*)d_in[35];

  // ---- workspace layout ----
  char* ws = (char*)d_ws;
  size_t off = 0;
  auto take = [&](size_t bytes) { size_t o = off; off += (bytes + 255) & ~(size_t)255; return o; };
  float* X   = (float*)(ws + take((size_t)MROWS * RELD * 4));
  float* Zf  = (float*)(ws + take((size_t)MROWS * RELD * 4));   // deltas / logits
  f16*   Xh  = (f16*)  (ws + take((size_t)MROWS * RELP * 2));
  f16*   BIG = (f16*)  (ws + take((size_t)MROWS * DFFP * 2));   // QKV+O / H / hidden
  f16*   Wf  = (f16*)  (ws + take((size_t)16785408 * 2));
  float* Bf  = (float*)(ws + take((size_t)15360 * 4));

  // ---- convert weights/biases to padded f16/f32 ----
  struct WS_ { int src; int N, K, Np, Kp; size_t woff; };
  struct BS_ { int src; int N, Np; size_t boff; };
  const size_t LW = 8146944;   // f16 elems per layer
  static const WS_ wspec[] = {
    {10, QKVN, RELD, QKVP, RELP, 0},
    {12, RELD, RELD, RELP, RELP, 2023424},
    {14, DFFN, RELD, DFFP, RELP, 2715648},
    {16, RELD, DFFN, RELP, DFFP, 5431296},
    {22, QKVN, RELD, QKVP, RELP, LW + 0},
    {24, RELD, RELD, RELP, RELP, LW + 2023424},
    {26, DFFN, RELD, DFFP, RELP, LW + 2715648},
    {28, RELD, DFFN, RELP, DFFP, LW + 5431296},
    {36, HIDN, RELD, HIDN, RELP, 16293888},
    {38, NRELN, HIDN, NRELP, HIDN, 16719872},
  };
  static const BS_ bspec[] = {
    {11, QKVN, QKVP, 0}, {13, RELD, RELP, 2432}, {15, DFFN, DFFP, 3264}, {17, RELD, RELP, 6528},
    {23, QKVN, QKVP, 7360}, {25, RELD, RELP, 9792}, {27, DFFN, DFFP, 10624}, {29, RELD, RELP, 13888},
    {37, HIDN, HIDN, 14720}, {39, NRELN, NRELP, 15232},
  };
  for (int i = 0; i < 10; ++i) {
    const WS_& w = wspec[i];
    long total = (long)w.Np * w.Kp;
    conv_w<<<dim3((total + 255) / 256), dim3(256), 0, stream>>>(
        (const float*)d_in[w.src], Wf + w.woff, w.N, w.K, w.Kp, total);
    const BS_& b = bspec[i];
    conv_b<<<dim3((b.Np + 255) / 256), dim3(256), 0, stream>>>(
        (const float*)d_in[b.src], Bf + b.boff, b.N, b.Np);
  }

  // ---- feature build ----
  build_rel<<<dim3(MROWS), dim3(256), 0, stream>>>(
      mse, hidx, hmask, tidx, tmask, htidx, htmask, d1, d2, disemb, X);

  const dim3 blk(256);
  const int gy = MROWS / 128;                      // 192
  f16* Of16 = BIG + (size_t)MROWS * QKVP;          // attn output region

  for (int L = 0; L < 2; ++L) {
    const size_t wb = (size_t)L * LW;
    const size_t bb = (size_t)L * 7360;
    const float* ln1g = (const float*)d_in[10 + L * 12 + 8];
    const float* ln1b = (const float*)d_in[10 + L * 12 + 9];
    const float* ln2g = (const float*)d_in[10 + L * 12 + 10];
    const float* ln2b = (const float*)d_in[10 + L * 12 + 11];

    conv_x<<<dim3(MROWS * RELP / 256), blk, 0, stream>>>(X, Xh);
    // QKV = Xh @ in_w^T + in_b   -> BIG f16 [M x 2432]
    gemm_wmma<<<dim3(QKVP / 64, gy), blk, 0, stream>>>(
        Xh, RELP, Wf + wb, RELP, Bf + bb, BIG, QKVP, nullptr, 0, QKVP, RELP, 0);
    attn_kernel<<<dim3(NRB * NHEADS), blk, 0, stream>>>(BIG, htmask, Of16);
    zero_pad_o<<<dim3(MROWS * 24 / 256), blk, 0, stream>>>(Of16);
    // Z = O @ out_w^T + out_b  (f32)
    gemm_wmma<<<dim3(RELP / 64, gy), blk, 0, stream>>>(
        Of16, RELP, Wf + wb + 2023424, RELP, Bf + bb + 2432,
        nullptr, 0, Zf, RELD, RELD, RELP, 0);
    ln_kernel<<<dim3(MROWS), blk, 0, stream>>>(X, Zf, ln1g, ln1b);
    conv_x<<<dim3(MROWS * RELP / 256), blk, 0, stream>>>(X, Xh);
    // H = relu(Xh @ lin1^T + lin1_b) -> BIG f16 [M x 3264]
    gemm_wmma<<<dim3(DFFP / 64, gy), blk, 0, stream>>>(
        Xh, RELP, Wf + wb + 2715648, RELP, Bf + bb + 3264,
        BIG, DFFP, nullptr, 0, DFFP, RELP, 1);
    // Z = H @ lin2^T + lin2_b (f32)
    gemm_wmma<<<dim3(RELP / 64, gy), blk, 0, stream>>>(
        BIG, DFFP, Wf + wb + 5431296, DFFP, Bf + bb + 6528,
        nullptr, 0, Zf, RELD, RELD, DFFP, 0);
    ln_kernel<<<dim3(MROWS), blk, 0, stream>>>(X, Zf, ln2g, ln2b);
  }

  // final norm
  ln_kernel<<<dim3(MROWS), blk, 0, stream>>>(X, nullptr, norm_g, norm_b);
  conv_x<<<dim3(MROWS * RELP / 256), blk, 0, stream>>>(X, Xh);
  // hidden = relu(Xh @ fp1^T + fp1_b) -> BIG f16 [M x 512]
  gemm_wmma<<<dim3(HIDN / 64, gy), blk, 0, stream>>>(
      Xh, RELP, Wf + 16293888, RELP, Bf + 14720, BIG, HIDN, nullptr, 0, HIDN, RELP, 1);
  // logits = hidden @ fp2^T + fp2_b -> Zf f32 [M x 128]
  gemm_wmma<<<dim3(NRELP / 64, gy), blk, 0, stream>>>(
      BIG, HIDN, Wf + 16719872, HIDN, Bf + 15232, nullptr, 0, Zf, NRELP, NRELP, HIDN, 0);
  // masked logsumexp over HT
  lse_kernel<<<dim3(NRB), dim3(128), 0, stream>>>(Zf, htmask, (float*)d_out);
}
